// ActTransNet_77309411328535
// MI455X (gfx1250) — matrix operand order repaired and checked by reading, not compiled, not run on hardware
//
#include <hip/hip_runtime.h>
#include <hip/hip_bf16.h>

#define B_SZ     4096
#define T_SZ     16
#define IN_DIM   2048
#define DIM      256
#define N_ACT    1000

typedef __attribute__((ext_vector_type(16))) __bf16 v16bf;
typedef __attribute__((ext_vector_type(8)))  float  v8f;
typedef __attribute__((ext_vector_type(4)))  float  f4;    // native vec4 for NT loads

union Frag16 { v16bf v; int4 q[2]; };
union BF4    { __hip_bfloat16 h[4]; uint2 u; };

// ---------------------------------------------------------------------------
// Kernel 1: mean over T (16 steps) + f32 -> bf16 convert.
// Inputs are read exactly once -> non-temporal loads to keep L2 for W_trans.
// grid.x covers B*IN_DIM/4 float4 lanes, grid.y selects precondition/effect.
// ---------------------------------------------------------------------------
__global__ __launch_bounds__(256) void pool_cvt(
    const float* __restrict__ pre, const float* __restrict__ eff,
    __hip_bfloat16* __restrict__ p_avg, __hip_bfloat16* __restrict__ e_avg)
{
    const float* src = blockIdx.y ? eff : pre;
    __hip_bfloat16* dst = blockIdx.y ? e_avg : p_avg;

    size_t g = (size_t)blockIdx.x * blockDim.x + threadIdx.x;   // float4 index
    size_t b = g >> 9;          // / (IN_DIM/4 = 512)
    size_t c = g & 511;

    const f4* base = (const f4*)src + b * (size_t)(T_SZ * 512) + c;
    f4 s = {0.f, 0.f, 0.f, 0.f};
#pragma unroll
    for (int t = 0; t < T_SZ; ++t) {
        f4 v = __builtin_nontemporal_load(base + (size_t)t * 512);
        s += v;
    }
    const float inv = 1.0f / (float)T_SZ;
    BF4 o;
    o.h[0] = __float2bfloat16(s.x * inv);
    o.h[1] = __float2bfloat16(s.y * inv);
    o.h[2] = __float2bfloat16(s.z * inv);
    o.h[3] = __float2bfloat16(s.w * inv);
    *(uint2*)(dst + 4 * g) = o.u;
}

// ---------------------------------------------------------------------------
// Kernel 2: f32 -> bf16 convert of the two [DIM, IN_DIM] weight matrices.
// ---------------------------------------------------------------------------
__global__ __launch_bounds__(256) void cvt_w(
    const float* __restrict__ Wp, const float* __restrict__ We,
    __hip_bfloat16* __restrict__ Wpb, __hip_bfloat16* __restrict__ Web)
{
    const float* src = blockIdx.y ? We : Wp;
    __hip_bfloat16* dst = blockIdx.y ? Web : Wpb;
    size_t g = (size_t)blockIdx.x * blockDim.x + threadIdx.x;   // float4 index
    float4 v = ((const float4*)src)[g];
    BF4 o;
    o.h[0] = __float2bfloat16(v.x);
    o.h[1] = __float2bfloat16(v.y);
    o.h[2] = __float2bfloat16(v.z);
    o.h[3] = __float2bfloat16(v.w);
    *(uint2*)(dst + 4 * g) = o.u;
}

// ---------------------------------------------------------------------------
// Kernel 3: WMMA GEMM  out[M,N] = A[M,K] @ W[N,K]^T + bias,  bf16 in / f32 out
// M=4096, K=2048, N=256. One wave -> 16x64 output strip (4 accumulators),
// double-buffered K-loop: loads for step k+32 are issued before the WMMAs
// consuming step k, so the pre-WMMA wait is loadcnt<=10 instead of 0.
// 8 waves per block; grid.x = 128 blocks; grid.y selects p / e problem.
// ---------------------------------------------------------------------------
__global__ __launch_bounds__(256) void gemm_wmma(
    const __hip_bfloat16* __restrict__ Apb, const __hip_bfloat16* __restrict__ Aeb,
    const __hip_bfloat16* __restrict__ Wpb, const __hip_bfloat16* __restrict__ Web,
    const float* __restrict__ bp, const float* __restrict__ be,
    float* __restrict__ outp, float* __restrict__ oute)
{
    const __hip_bfloat16* A    = blockIdx.y ? Aeb : Apb;
    const __hip_bfloat16* W    = blockIdx.y ? Web : Wpb;
    const float*          bias = blockIdx.y ? be  : bp;
    float*                out  = blockIdx.y ? oute : outp;

    const int lane = threadIdx.x & 31;
    const int wid  = blockIdx.x * 8 + (threadIdx.x >> 5);   // 0..1023
    const int nTile = (wid & 3) * 64;                       // 0,64,128,192
    const int mTile = (wid >> 2) * 16;                      // 0..4080

    // A-fragment addressing (contiguous k order, same as B): row = lane&15,
    // K-halves: lanes 0-15 hold K [0..15], lanes 16-31 [16..31]
    const int koff = (lane >> 4) * 16;
    // B-fragment addressing: col = lane&15, lanes 0-15 hold K 0..15,
    // lanes 16-31 hold K 16..31 (contiguous 32 bytes of the weight row).
    const int kbase = (lane >> 4) * 16;

    const __hip_bfloat16* arowp = A + (size_t)(mTile + (lane & 15)) * IN_DIM;
    const __hip_bfloat16* wrow0 = W + (size_t)(nTile + (lane & 15)) * IN_DIM + kbase;

    v8f acc[4] = {};

    Frag16 aCur, bCur[4];
    aCur.q[0] = *(const int4*)(arowp + koff);
    aCur.q[1] = *(const int4*)(arowp + koff + 8);
#pragma unroll
    for (int t = 0; t < 4; ++t) {
        const __hip_bfloat16* wp = wrow0 + (size_t)t * 16 * IN_DIM;
        bCur[t].q[0] = *(const int4*)(wp);
        bCur[t].q[1] = *(const int4*)(wp + 8);
    }

    for (int k = 0; k < IN_DIM - 32; k += 32) {
        const int kn = k + 32;
        // ---- issue next-step loads first (overlap with WMMAs below) ----
        Frag16 aNxt, bNxt[4];
        aNxt.q[0] = *(const int4*)(arowp + kn + koff);
        aNxt.q[1] = *(const int4*)(arowp + kn + koff + 8);
#pragma unroll
        for (int t = 0; t < 4; ++t) {
            const __hip_bfloat16* wp = wrow0 + (size_t)t * 16 * IN_DIM + kn;
            bNxt[t].q[0] = *(const int4*)(wp);
            bNxt[t].q[1] = *(const int4*)(wp + 8);
        }
        // ---- matrix math on current fragments ----
#pragma unroll
        for (int t = 0; t < 4; ++t)
            acc[t] = __builtin_amdgcn_wmma_f32_16x16x32_bf16(
                false, aCur.v, false, bCur[t].v, (short)0, acc[t], false, false);
        // ---- rotate buffers ----
        aCur = aNxt;
#pragma unroll
        for (int t = 0; t < 4; ++t) bCur[t] = bNxt[t];
    }
    // epilogue K-step
#pragma unroll
    for (int t = 0; t < 4; ++t)
        acc[t] = __builtin_amdgcn_wmma_f32_16x16x32_bf16(
            false, aCur.v, false, bCur[t].v, (short)0, acc[t], false, false);

    // D layout: VGPR r -> row mTile + (lane>>4)*8 + r, col = lane&15 (+ tile n)
    const int rbase = mTile + (lane >> 4) * 8;
#pragma unroll
    for (int t = 0; t < 4; ++t) {
        const int n  = nTile + t * 16 + (lane & 15);
        const float bv = bias[n];
#pragma unroll
        for (int r = 0; r < 8; ++r)
            out[(size_t)(rbase + r) * DIM + n] = acc[t][r] + bv;
    }
}

// ---------------------------------------------------------------------------
// Kernel 4: p_transformed[b] = W_trans[action[b]] @ p_embed[b]
// One block (8 waves) per sample; vector in LDS; coalesced float4 row reads;
// wave32 shuffle reduction.
// ---------------------------------------------------------------------------
__global__ __launch_bounds__(256) void gather_matvec(
    const float* __restrict__ Wt, const int* __restrict__ action,
    const float* __restrict__ vin, float* __restrict__ out)
{
    __shared__ float4 vs[DIM / 4];
    const int b   = blockIdx.x;
    const int tid = threadIdx.x;

    if (tid < DIM / 4)
        vs[tid] = ((const float4*)(vin + (size_t)b * DIM))[tid];
    __syncthreads();

    const float* M = Wt + (size_t)action[b] * (DIM * DIM);
    const int wave = tid >> 5, lane = tid & 31;

    for (int r = 0; r < 32; ++r) {
        const int row = wave * 32 + r;
        const float4* mrow = (const float4*)(M + (size_t)row * DIM);
        float4 m0 = mrow[lane];
        float4 m1 = mrow[lane + 32];
        float4 v0 = vs[lane];
        float4 v1 = vs[lane + 32];
        float acc = m0.x * v0.x + m0.y * v0.y + m0.z * v0.z + m0.w * v0.w
                  + m1.x * v1.x + m1.y * v1.y + m1.z * v1.z + m1.w * v1.w;
#pragma unroll
        for (int off = 16; off > 0; off >>= 1)
            acc += __shfl_down(acc, off, 32);
        if (lane == 0)
            out[(size_t)b * DIM + row] = acc;
    }
}

// ---------------------------------------------------------------------------
extern "C" void kernel_launch(void* const* d_in, const int* in_sizes, int n_in,
                              void* d_out, int out_size, void* d_ws, size_t ws_size,
                              hipStream_t stream) {
    (void)in_sizes; (void)n_in; (void)out_size; (void)ws_size;

    const float* pre    = (const float*)d_in[0];
    const float* eff    = (const float*)d_in[1];
    const int*   action = (const int*)  d_in[2];
    const float* Wp     = (const float*)d_in[3];
    const float* bp     = (const float*)d_in[4];
    const float* We     = (const float*)d_in[5];
    const float* be     = (const float*)d_in[6];
    const float* Wt     = (const float*)d_in[7];
    float*       out    = (float*)d_out;   // [B*DIM) p_transformed | [B*DIM) e_embed

    char* ws = (char*)d_ws;
    __hip_bfloat16* p_avg  = (__hip_bfloat16*)(ws);                        // 16 MB
    __hip_bfloat16* e_avg  = (__hip_bfloat16*)(ws + (16u << 20));          // 16 MB
    __hip_bfloat16* Wpb    = (__hip_bfloat16*)(ws + (32u << 20));          //  1 MB
    __hip_bfloat16* Web    = (__hip_bfloat16*)(ws + (33u << 20));          //  1 MB
    float*          p_emb  = (float*)        (ws + (34u << 20));           //  4 MB

    // 1) mean-pool + bf16 convert: B*IN_DIM/4 = 2M float4 per tensor
    pool_cvt<<<dim3((B_SZ * IN_DIM / 4) / 256, 2), 256, 0, stream>>>(
        pre, eff, p_avg, e_avg);

    // 2) weight convert: DIM*IN_DIM/4 = 128K float4 per matrix
    cvt_w<<<dim3((DIM * IN_DIM / 4) / 256, 2), 256, 0, stream>>>(
        Wp, We, Wpb, Web);

    // 3) WMMA GEMMs: 1024 waves per problem / 8 waves per block = 128 blocks
    gemm_wmma<<<dim3(128, 2), 256, 0, stream>>>(
        p_avg, e_avg, Wpb, Web, bp, be, p_emb, out + (size_t)B_SZ * DIM);

    // 4) gather + batched matvec -> p_transformed
    gather_matvec<<<B_SZ, 256, 0, stream>>>(Wt, action, p_emb, out);
}
